// MultiHeadSelfAttentionBlock_75943611728622
// MI455X (gfx1250) — matrix operand order repaired and checked by reading, not compiled, not run on hardware
//
#include <hip/hip_runtime.h>

// MI455X (gfx1250) multi-head self-attention block, bf16 WMMA + fp32 accumulate.
// E=1024, H=16, HD=64, B=2, T=2048, MAX_REL=128.
// Weight GEMMs stage 64x64 bf16 weight tiles through LDS with the Tensor Data
// Mover (tensor_load_to_lds, TENSORcnt), double-buffered against WMMA compute.

typedef __bf16 bf16;
typedef __attribute__((ext_vector_type(16))) __bf16 v16bf;
typedef __attribute__((ext_vector_type(8)))  __bf16 v8bf;
typedef __attribute__((ext_vector_type(8)))  float  v8f;
typedef __attribute__((ext_vector_type(4)))  unsigned int v4u;
typedef __attribute__((ext_vector_type(8)))  int v8i;
typedef __attribute__((ext_vector_type(4)))  int v4i;

#define E_DIM 1024
#define T_DIM 2048
#define H_DIM 16
#define HD_DIM 64
#define NTOK 4096   // B*T

#if __has_builtin(__builtin_amdgcn_tensor_load_to_lds)
#define HAVE_TDM 1
#else
#define HAVE_TDM 0
#endif

// ---------------- WMMA fragment helpers -----------------------------------
// A fragment (16x32 bf16, row-major source, leading dim ld):
// lane<16 : M=lane,    K = {0..7, 16..23} ; lane>=16: M=lane-16, K = {8..15, 24..31}
__device__ inline v16bf load_a_frag(const bf16* __restrict__ base, int ld) {
  int lane = threadIdx.x & 31;
  int r = lane & 15;
  int kb = (lane < 16) ? 0 : 8;
  const bf16* p = base + (size_t)r * ld + kb;
  v8bf lo = *(const v8bf*)(p);
  v8bf hi = *(const v8bf*)(p + 16);
  v16bf a;
#pragma unroll
  for (int i = 0; i < 8; ++i) { a[i] = lo[i]; a[i + 8] = hi[i]; }
  return a;
}

// B fragment (32x16 bf16). Column n of B = row n of the row-major source.
// lanes 0-15: N=lane, K={0..7,16..23} ; lanes 16-31: N=lane-16, K={8..15,24..31}.
__device__ inline v16bf load_b_frag(const bf16* __restrict__ base, int ld) {
  int lane = threadIdx.x & 31;
  int n = lane & 15;
  int kb = (lane < 16) ? 0 : 8;
  const bf16* p = base + (size_t)n * ld + kb;
  v8bf lo = *(const v8bf*)(p);
  v8bf hi = *(const v8bf*)(p + 16);
  v16bf b;
#pragma unroll
  for (int i = 0; i < 8; ++i) { b[i] = lo[i]; b[i + 8] = hi[i]; }
  return b;
}

__device__ inline v8f wmma_bf16(v16bf a, v16bf b, v8f c) {
  return __builtin_amdgcn_wmma_f32_16x16x32_bf16(false, a, false, b, (short)0, c,
                                                 false, false);
}

// ---------------- Tensor Data Mover: 2-D bf16 tile -> LDS ------------------
// D# per CDNA5 ISA:
//   group0: count=1, lds_addr, global_addr(57b), type=2
//   group1: data_size=1 (2 bytes), tensor_dim0/1, tile_dim0/1, dim0 stride
// Issued by one wave per block (per-wave DMA, EXEC-independent, TENSORcnt).
__device__ inline void tdm_load_tile_2d(unsigned lds_off, const void* gaddr,
                                        unsigned tensor_d0, unsigned tensor_d1,
                                        unsigned stride0, unsigned tile_d0,
                                        unsigned tile_d1) {
#if HAVE_TDM
  unsigned long long ga = (unsigned long long)(size_t)gaddr;
  v4u g0;
  g0[0] = 1u;                                         // count=1, no gather
  g0[1] = lds_off;                                    // D#.lds_addr (bytes)
  g0[2] = (unsigned)(ga & 0xffffffffu);               // global_addr[31:0]
  g0[3] = (unsigned)((ga >> 32) & 0x01ffffffu)        // global_addr[56:32]
          | (2u << 30);                               // type = 2 (image)
  v8i g1;
  g1[0] = (int)(1u << 16);                            // data_size=1 -> 2 bytes
  g1[1] = (int)((tensor_d0 & 0xffffu) << 16);         // tensor_dim0[15:0]
  g1[2] = (int)(((tensor_d0 >> 16) & 0xffffu) |       // tensor_dim0[31:16]
                ((tensor_d1 & 0xffffu) << 16));       // tensor_dim1[15:0]
  g1[3] = (int)(((tensor_d1 >> 16) & 0xffffu) |       // tensor_dim1[31:16]
                ((tile_d0 & 0xffffu) << 16));         // tile_dim0
  g1[4] = (int)(tile_d1 & 0xffffu);                   // tile_dim1 (tile_dim2=0)
  g1[5] = (int)stride0;                               // tensor_dim0_stride[31:0]
  g1[6] = 0;                                          // stride hi + dim1_stride lo
  g1[7] = 0;
  v4i z4 = {0, 0, 0, 0};
  v8i z8 = {0, 0, 0, 0, 0, 0, 0, 0};
  __builtin_amdgcn_tensor_load_to_lds(g0, g1, z4, z4, z8, 0);
#else
  (void)lds_off; (void)gaddr; (void)tensor_d0; (void)tensor_d1;
  (void)stride0; (void)tile_d0; (void)tile_d1;
#endif
}

__device__ inline void wait_tensorcnt0() {
#if __has_builtin(__builtin_amdgcn_s_wait_tensorcnt)
  __builtin_amdgcn_s_wait_tensorcnt(0);
#else
  asm volatile("s_wait_tensorcnt 0x0" ::: "memory");
#endif
}

// Fallback cooperative copy (compiled only if TDM builtin is unavailable).
__device__ inline void coop_copy_tile(bf16* dst, const bf16* src, int ld) {
#if !HAVE_TDM
  // 64 rows x 64 bf16 = 8KB, 256 threads x 32B
  int t = threadIdx.x;
#pragma unroll
  for (int i = 0; i < 2; ++i) {
    int c = t + i * 256;          // 0..511 chunks of 16B
    int row = c >> 3;
    int col = (c & 7) * 8;
    *(v8bf*)(dst + row * 64 + col) = *(const v8bf*)(src + (size_t)row * ld + col);
  }
#else
  (void)dst; (void)src; (void)ld;
#endif
}

// ---------------- 1. LayerNorm -> bf16 ------------------------------------
__global__ void ln_kernel(const float* __restrict__ x, const float* __restrict__ gamma,
                          const float* __restrict__ beta, bf16* __restrict__ out) {
  const int row = blockIdx.x;                       // 0..4095
  const float* xr = x + (size_t)row * E_DIM;
  float4 v = ((const float4*)xr)[threadIdx.x];      // 256 thr * 4 = 1024
  float s  = v.x + v.y + v.z + v.w;
  float ss = v.x * v.x + v.y * v.y + v.z * v.z + v.w * v.w;
#pragma unroll
  for (int m = 1; m < 32; m <<= 1) {
    s  += __shfl_xor(s,  m, 32);
    ss += __shfl_xor(ss, m, 32);
  }
  __shared__ float red0[8], red1[8];
  int wid = threadIdx.x >> 5, lane = threadIdx.x & 31;
  if (lane == 0) { red0[wid] = s; red1[wid] = ss; }
  __syncthreads();
  float ts = 0.f, tss = 0.f;
#pragma unroll
  for (int i = 0; i < 8; ++i) { ts += red0[i]; tss += red1[i]; }
  float mu   = ts * (1.0f / E_DIM);
  float var  = tss * (1.0f / E_DIM) - mu * mu;
  float rstd = rsqrtf(var + 1e-5f);
  int c0 = threadIdx.x * 4;
  bf16* orow = out + (size_t)row * E_DIM + c0;
  float vv[4] = {v.x, v.y, v.z, v.w};
#pragma unroll
  for (int i = 0; i < 4; ++i)
    orow[i] = (bf16)((vv[i] - mu) * rstd * gamma[c0 + i] + beta[c0 + i]);
}

// ---------------- 2. fp32 -> bf16 weight convert ---------------------------
__global__ void cvt_bf16_kernel(const float* __restrict__ src, bf16* __restrict__ dst,
                                int n) {
  int i = blockIdx.x * blockDim.x + threadIdx.x;
  if (i < n) dst[i] = (bf16)src[i];
}

// ---------------- 3. QKV projection GEMMs (TDM-staged weights) -------------
// C[m,n] = sum_k h[m,k] * W[n,k] + bias[n]   (NT GEMM)
// q,k written as [bh][t][d] bf16 ; v written transposed [bh][d][t] bf16.
__global__ void qkv_gemm_kernel(const bf16* __restrict__ hbf,
                                const bf16* __restrict__ Wq, const bf16* __restrict__ Wk,
                                const bf16* __restrict__ Wv,
                                const float* __restrict__ bq, const float* __restrict__ bk,
                                const float* __restrict__ bv,
                                bf16* __restrict__ qout, bf16* __restrict__ kout,
                                bf16* __restrict__ vT) {
  const int which = blockIdx.z;
  const bf16* W = (which == 0) ? Wq : (which == 1) ? Wk : Wv;
  const float* bias = (which == 0) ? bq : (which == 1) ? bk : bv;
  bf16* qk = (which == 0) ? qout : kout;

  __shared__ __attribute__((aligned(32))) bf16 wtile[2][64][64];   // 16 KiB

  const int wid  = threadIdx.x >> 5;
  const int lane = threadIdx.x & 31;
  const int nblk = blockIdx.x * 64;
  const int m0 = blockIdx.y * 128 + (wid >> 1) * 32;
  const int nl = (wid & 1) * 32;            // wave's n offset inside block tile

  const unsigned ldsb[2] = {(unsigned)(size_t)&wtile[0][0][0],
                            (unsigned)(size_t)&wtile[1][0][0]};

  // prologue: DMA first 64x64 weight tile (rows n, cols k) into buffer 0
  if (HAVE_TDM) {
    if (wid == 0)
      tdm_load_tile_2d(ldsb[0], W + (size_t)nblk * E_DIM, E_DIM, 64, E_DIM, 64, 64);
  } else {
    coop_copy_tile(&wtile[0][0][0], W + (size_t)nblk * E_DIM, E_DIM);
    __syncthreads();
  }

  v8f c00 = {}, c01 = {}, c10 = {}, c11 = {};
  int cur = 0;
  for (int kc = 0; kc < E_DIM; kc += 64) {
    if (HAVE_TDM) {
      if (wid == 0) wait_tensorcnt0();
      __syncthreads();                      // buf[cur] ready; buf[cur^1] free
      if (wid == 0 && kc + 64 < E_DIM)
        tdm_load_tile_2d(ldsb[cur ^ 1], W + (size_t)nblk * E_DIM + kc + 64,
                         E_DIM, 64, E_DIM, 64, 64);
    }
    const bf16* wt = &wtile[cur][0][0];
#pragma unroll
    for (int ks = 0; ks < 64; ks += 32) {
      v16bf a0 = load_a_frag(hbf + (size_t)m0 * E_DIM + kc + ks, E_DIM);
      v16bf a1 = load_a_frag(hbf + (size_t)(m0 + 16) * E_DIM + kc + ks, E_DIM);
      v16bf b0 = load_b_frag(wt + (nl + 0) * 64 + ks, 64);
      v16bf b1 = load_b_frag(wt + (nl + 16) * 64 + ks, 64);
      c00 = wmma_bf16(a0, b0, c00);
      c01 = wmma_bf16(a0, b1, c01);
      c10 = wmma_bf16(a1, b0, c10);
      c11 = wmma_bf16(a1, b1, c11);
    }
    if (!HAVE_TDM && kc + 64 < E_DIM) {
      __syncthreads();
      coop_copy_tile(&wtile[cur ^ 1][0][0], W + (size_t)nblk * E_DIM + kc + 64, E_DIM);
      __syncthreads();
    }
    cur ^= 1;
  }

  const int mhalf = (lane < 16) ? 0 : 8;
  const int nloc  = lane & 15;
  const int n0 = nblk + nl;
  v8f tiles[4] = {c00, c01, c10, c11};
#pragma unroll
  for (int t = 0; t < 4; ++t) {
    int mb = m0 + (t >> 1) * 16;
    int nb = n0 + (t & 1) * 16;
#pragma unroll
    for (int r = 0; r < 8; ++r) {
      int m = mb + mhalf + r;            // token index
      int n = nb + nloc;                 // embedding col
      float val = tiles[t][r] + bias[n];
      int b = m >> 11, tt = m & (T_DIM - 1);
      int hh = n >> 6, d = n & (HD_DIM - 1);
      int bh = b * H_DIM + hh;
      if (which == 2)
        vT[((size_t)bh * HD_DIM + d) * T_DIM + tt] = (bf16)val;
      else
        qk[((size_t)bh * T_DIM + tt) * HD_DIM + d] = (bf16)val;
    }
  }
}

// ---------------- 4. Flash attention ---------------------------------------
// One block = 8 waves; each wave owns 16 query rows of one (b,h).
__global__ void attn_kernel(const bf16* __restrict__ qbf, const bf16* __restrict__ kbf,
                            const bf16* __restrict__ vT, const float* __restrict__ rel_emb,
                            const unsigned char* __restrict__ pad,
                            bf16* __restrict__ obf) {
  const int bh = blockIdx.y;                 // 0..31
  const int b = bh >> 4, h = bh & 15;
  const int wid  = threadIdx.x >> 5;
  const int lane = threadIdx.x & 31;
  const int q0 = blockIdx.x * 128 + wid * 16;

  const bf16* Q = qbf + (size_t)bh * T_DIM * HD_DIM;
  const bf16* K = kbf + (size_t)bh * T_DIM * HD_DIM;
  const bf16* V = vT + (size_t)bh * HD_DIM * T_DIM;

  __shared__ __attribute__((aligned(32))) bf16 pbuf[8][16][32];

  v16bf aq0 = load_a_frag(Q + (size_t)q0 * HD_DIM + 0, HD_DIM);
  v16bf aq1 = load_a_frag(Q + (size_t)q0 * HD_DIM + 32, HD_DIM);

  v8f o0 = {}, o1 = {}, o2 = {}, o3 = {};
  float mrow[8], lrow[8];
#pragma unroll
  for (int r = 0; r < 8; ++r) { mrow[r] = -1e30f; lrow[r] = 0.f; }

  const int nloc  = lane & 15;
  const int mhalf = (lane < 16) ? 0 : 8;

  for (int kb = 0; kb < T_DIM; kb += 32) {
    float s0[8], s1[8];
    {
      int t_k = kb + nloc;
      v16bf bk0 = load_b_frag(K + (size_t)kb * HD_DIM + 0, HD_DIM);
      v16bf bk1 = load_b_frag(K + (size_t)kb * HD_DIM + 32, HD_DIM);
      v8f s = {};
      s = wmma_bf16(aq0, bk0, s);
      s = wmma_bf16(aq1, bk1, s);
      bool ok = pad[b * T_DIM + t_k] != 0;
#pragma unroll
      for (int r = 0; r < 8; ++r) {
        int t_q = q0 + mhalf + r;
        int rel = t_k - t_q;
        rel = rel < -128 ? -128 : (rel > 128 ? 128 : rel);
        float bias = rel_emb[(rel + 128) * H_DIM + h];
        s0[r] = ok ? (s[r] * 0.125f + bias) : -1e30f;
      }
    }
    {
      int t_k = kb + 16 + nloc;
      v16bf bk0 = load_b_frag(K + (size_t)(kb + 16) * HD_DIM + 0, HD_DIM);
      v16bf bk1 = load_b_frag(K + (size_t)(kb + 16) * HD_DIM + 32, HD_DIM);
      v8f s = {};
      s = wmma_bf16(aq0, bk0, s);
      s = wmma_bf16(aq1, bk1, s);
      bool ok = pad[b * T_DIM + t_k] != 0;
#pragma unroll
      for (int r = 0; r < 8; ++r) {
        int t_q = q0 + mhalf + r;
        int rel = t_k - t_q;
        rel = rel < -128 ? -128 : (rel > 128 ? 128 : rel);
        float bias = rel_emb[(rel + 128) * H_DIM + h];
        s1[r] = ok ? (s[r] * 0.125f + bias) : -1e30f;
      }
    }
    // online softmax (rows live in the 16-lane halves of the C layout)
#pragma unroll
    for (int r = 0; r < 8; ++r) {
      float tm = fmaxf(s0[r], s1[r]);
#pragma unroll
      for (int msk = 1; msk < 16; msk <<= 1) tm = fmaxf(tm, __shfl_xor(tm, msk, 32));
      float nm = fmaxf(mrow[r], tm);
      float corr = __expf(mrow[r] - nm);
      float p0 = __expf(s0[r] - nm);
      float p1 = __expf(s1[r] - nm);
      float rs = p0 + p1;
#pragma unroll
      for (int msk = 1; msk < 16; msk <<= 1) rs += __shfl_xor(rs, msk, 32);
      lrow[r] = lrow[r] * corr + rs;
      mrow[r] = nm;
      o0[r] *= corr; o1[r] *= corr; o2[r] *= corr; o3[r] *= corr;
      int m = mhalf + r;
      pbuf[wid][m][nloc]      = (bf16)p0;   // C-layout -> row-major P in LDS
      pbuf[wid][m][nloc + 16] = (bf16)p1;
    }
    // per-wave LDS is in order; wait for stores before the A-layout reload
    asm volatile("s_wait_dscnt 0x0" ::: "memory");
    v16bf ap  = load_a_frag(&pbuf[wid][0][0], 32);   // P as A fragment (16x32)
    v16bf bv0 = load_b_frag(V + (size_t)0  * T_DIM + kb, T_DIM);
    v16bf bv1 = load_b_frag(V + (size_t)16 * T_DIM + kb, T_DIM);
    v16bf bv2 = load_b_frag(V + (size_t)32 * T_DIM + kb, T_DIM);
    v16bf bv3 = load_b_frag(V + (size_t)48 * T_DIM + kb, T_DIM);
    o0 = wmma_bf16(ap, bv0, o0);
    o1 = wmma_bf16(ap, bv1, o1);
    o2 = wmma_bf16(ap, bv2, o2);
    o3 = wmma_bf16(ap, bv3, o3);
  }

#pragma unroll
  for (int r = 0; r < 8; ++r) {
    float inv = 1.0f / lrow[r];
    int t_q = q0 + mhalf + r;
    size_t row = (size_t)b * T_DIM + t_q;
    int dbase = h * HD_DIM + nloc;
    obf[row * E_DIM + dbase + 0]  = (bf16)(o0[r] * inv);
    obf[row * E_DIM + dbase + 16] = (bf16)(o1[r] * inv);
    obf[row * E_DIM + dbase + 32] = (bf16)(o2[r] * inv);
    obf[row * E_DIM + dbase + 48] = (bf16)(o3[r] * inv);
  }
}

// ---------------- 5. Output projection + bias + residual -------------------
__global__ void out_gemm_kernel(const bf16* __restrict__ obf, const bf16* __restrict__ Wo,
                                const float* __restrict__ bo, const float* __restrict__ x,
                                float* __restrict__ out) {
  __shared__ __attribute__((aligned(32))) bf16 wtile[2][64][64];

  const int wid  = threadIdx.x >> 5;
  const int lane = threadIdx.x & 31;
  const int nblk = blockIdx.x * 64;
  const int m0 = blockIdx.y * 128 + (wid >> 1) * 32;
  const int nl = (wid & 1) * 32;

  const unsigned ldsb[2] = {(unsigned)(size_t)&wtile[0][0][0],
                            (unsigned)(size_t)&wtile[1][0][0]};

  if (HAVE_TDM) {
    if (wid == 0)
      tdm_load_tile_2d(ldsb[0], Wo + (size_t)nblk * E_DIM, E_DIM, 64, E_DIM, 64, 64);
  } else {
    coop_copy_tile(&wtile[0][0][0], Wo + (size_t)nblk * E_DIM, E_DIM);
    __syncthreads();
  }

  v8f c00 = {}, c01 = {}, c10 = {}, c11 = {};
  int cur = 0;
  for (int kc = 0; kc < E_DIM; kc += 64) {
    if (HAVE_TDM) {
      if (wid == 0) wait_tensorcnt0();
      __syncthreads();
      if (wid == 0 && kc + 64 < E_DIM)
        tdm_load_tile_2d(ldsb[cur ^ 1], Wo + (size_t)nblk * E_DIM + kc + 64,
                         E_DIM, 64, E_DIM, 64, 64);
    }
    const bf16* wt = &wtile[cur][0][0];
#pragma unroll
    for (int ks = 0; ks < 64; ks += 32) {
      v16bf a0 = load_a_frag(obf + (size_t)m0 * E_DIM + kc + ks, E_DIM);
      v16bf a1 = load_a_frag(obf + (size_t)(m0 + 16) * E_DIM + kc + ks, E_DIM);
      v16bf b0 = load_b_frag(wt + (nl + 0) * 64 + ks, 64);
      v16bf b1 = load_b_frag(wt + (nl + 16) * 64 + ks, 64);
      c00 = wmma_bf16(a0, b0, c00);
      c01 = wmma_bf16(a0, b1, c01);
      c10 = wmma_bf16(a1, b0, c10);
      c11 = wmma_bf16(a1, b1, c11);
    }
    if (!HAVE_TDM && kc + 64 < E_DIM) {
      __syncthreads();
      coop_copy_tile(&wtile[cur ^ 1][0][0], Wo + (size_t)nblk * E_DIM + kc + 64, E_DIM);
      __syncthreads();
    }
    cur ^= 1;
  }

  const int mhalf = (lane < 16) ? 0 : 8;
  const int nloc  = lane & 15;
  const int n0 = nblk + nl;
  v8f tiles[4] = {c00, c01, c10, c11};
#pragma unroll
  for (int t = 0; t < 4; ++t) {
    int mb = m0 + (t >> 1) * 16;
    int nb = n0 + (t & 1) * 16;
#pragma unroll
    for (int r = 0; r < 8; ++r) {
      int m = mb + mhalf + r;
      int n = nb + nloc;
      size_t idx = (size_t)m * E_DIM + n;
      out[idx] = tiles[t][r] + bo[n] + x[idx];
    }
  }
}

// ---------------- launcher -------------------------------------------------
extern "C" void kernel_launch(void* const* d_in, const int* in_sizes, int n_in,
                              void* d_out, int out_size, void* d_ws, size_t ws_size,
                              hipStream_t stream) {
  (void)in_sizes; (void)n_in; (void)out_size; (void)ws_size;
  const float* x     = (const float*)d_in[0];
  const unsigned char* pad = (const unsigned char*)d_in[1];   // bool mask (B,T)
  const float* gamma = (const float*)d_in[2];
  const float* beta  = (const float*)d_in[3];
  const float* Wq    = (const float*)d_in[4];
  const float* bq    = (const float*)d_in[5];
  const float* Wk    = (const float*)d_in[6];
  const float* bk    = (const float*)d_in[7];
  const float* Wv    = (const float*)d_in[8];
  const float* bv    = (const float*)d_in[9];
  const float* Wo    = (const float*)d_in[10];
  const float* bo    = (const float*)d_in[11];
  const float* rel   = (const float*)d_in[12];
  float* out = (float*)d_out;

  char* ws = (char*)d_ws;
  bf16* hbf = (bf16*)(ws + (0  << 20));  // 8 MiB  [4096,1024]
  bf16* wqb = (bf16*)(ws + (8  << 20));  // 2 MiB
  bf16* wkb = (bf16*)(ws + (10 << 20));  // 2 MiB
  bf16* wvb = (bf16*)(ws + (12 << 20));  // 2 MiB
  bf16* wob = (bf16*)(ws + (14 << 20));  // 2 MiB
  bf16* qbf = (bf16*)(ws + (16 << 20));  // 8 MiB  [bh][t][d]
  bf16* kbf = (bf16*)(ws + (24 << 20));  // 8 MiB  [bh][t][d]
  bf16* vT  = (bf16*)(ws + (32 << 20));  // 8 MiB  [bh][d][t]
  bf16* obf = (bf16*)(ws + (40 << 20));  // 8 MiB  [token][e]

  const int NW = E_DIM * E_DIM;  // 1M weight elems

  ln_kernel<<<NTOK, 256, 0, stream>>>(x, gamma, beta, hbf);
  cvt_bf16_kernel<<<(NW + 255) / 256, 256, 0, stream>>>(Wq, wqb, NW);
  cvt_bf16_kernel<<<(NW + 255) / 256, 256, 0, stream>>>(Wk, wkb, NW);
  cvt_bf16_kernel<<<(NW + 255) / 256, 256, 0, stream>>>(Wv, wvb, NW);
  cvt_bf16_kernel<<<(NW + 255) / 256, 256, 0, stream>>>(Wo, wob, NW);
  qkv_gemm_kernel<<<dim3(16, 32, 3), 256, 0, stream>>>(hbf, wqb, wkb, wvb,
                                                       bq, bk, bv, qbf, kbf, vT);
  attn_kernel<<<dim3(16, 32), 256, 0, stream>>>(qbf, kbf, vT, rel, pad, obf);
  out_gemm_kernel<<<dim3(16, 32), 256, 0, stream>>>(obf, wob, bo, x, out);
}
